// GraphAttentionLayer_41369124995416
// MI455X (gfx1250) — matrix operand order repaired and checked by reading, not compiled, not run on hardware
//
#include <hip/hip_runtime.h>
#include <hip/hip_bf16.h>

typedef __attribute__((ext_vector_type(16))) _Float16 v16h;
typedef __attribute__((ext_vector_type(8)))  float    v8f;
typedef __attribute__((ext_vector_type(4)))  int      v4i;

constexpr int kB = 8;
constexpr int kN = 2048;
constexpr int kF = 64;                       // F_IN == F_OUT
constexpr float kAlpha = 0.2f;
constexpr float kLog2e = 1.44269504088896340736f;
constexpr float kNegFill2 = -9.0e15f * 1.44269504088896340736f; // pre-scaled by log2(e)
constexpr int kSplit  = 4;                   // waves cooperating on one 16-row block
constexpr int kJChunk = kN / kSplit;         // 512 columns per wave

// ---------------------------------------------------------------------------
// Kernel 1: h = x @ W^T, stored TRANSPOSED as f16: hT[b][o][n] (contiguous K
// for the WMMA B operand). src/dst attention logits pre-scaled by log2(e) so
// the hot loop can use raw v_exp_f32 (2^x) with no extra multiply.
// ---------------------------------------------------------------------------
__global__ __launch_bounds__(256) void gat_proj_kernel(
    const float* __restrict__ x,      // [B,N,64]
    const float* __restrict__ W,      // [64,64]
    const float* __restrict__ a,      // [128]
    _Float16* __restrict__ hT,        // [B,64,N]
    float* __restrict__ srcv,         // [B,N]   (log2-scaled)
    float* __restrict__ dstv)         // [B,N]   (log2-scaled)
{
    __shared__ float sW[kF * kF];     // 16 KB
    __shared__ float sa[2 * kF];

    const int tid = threadIdx.x;
    for (int t = tid; t < kF * kF / 4; t += 256)
        ((float4*)sW)[t] = ((const float4*)W)[t];
    if (tid < 32)
        ((float4*)sa)[tid] = ((const float4*)a)[tid];
    __syncthreads();

    const int idx = blockIdx.x * 256 + tid;       // 0 .. B*N-1
    const int b   = idx >> 11;
    const int n   = idx & (kN - 1);

    const float* xr = x + (long)idx * kF;
    float xv[kF];
#pragma unroll
    for (int f4 = 0; f4 < kF / 4; ++f4)
        ((float4*)xv)[f4] = ((const float4*)xr)[f4];

    float s = 0.f, d = 0.f;
#pragma unroll 4
    for (int o = 0; o < kF; ++o) {
        float acc = 0.f;
#pragma unroll
        for (int f = 0; f < kF; ++f)
            acc = fmaf(xv[f], sW[o * kF + f], acc);
        hT[((long)b * kF + o) * kN + n] = (_Float16)acc;
        s = fmaf(acc, sa[o], s);
        d = fmaf(acc, sa[kF + o], d);
    }
    srcv[idx] = s * kLog2e;
    dstv[idx] = d * kLog2e;
}

// ---------------------------------------------------------------------------
// Kernel 2: fused masked-softmax attention + (attention @ h) via WMMA.
// 256-thread block = 2 row-blocks x 4 j-split waves. Each wave runs an online
// (flash) softmax over its 512-column slice; partials merged in LDS.
//
// A (16-bit, 16x32): M = lane%16; lanes 0-15 own K {0-7,8-15},
//                                 lanes 16-31 own K {16-23,24-31}.
// B (16-bit, 32x16): Ncol = lane%16; lanes 0-15 K 0..15, lanes 16-31 K 16..31.
// C/D (f32 16x16):   element r <-> row r + 8*(lane>=16), col = lane%16.
// ---------------------------------------------------------------------------
__global__ __launch_bounds__(256) void gat_attn_kernel(
    const int* __restrict__ adj,      // [B,N,N]
    const _Float16* __restrict__ hT,  // [B,64,N]
    const float* __restrict__ srcv,   // [B,N] log2-scaled
    const float* __restrict__ dstv,   // [B,N] log2-scaled
    float* __restrict__ out)          // [B,N,64]
{
    __shared__ float s_dst[kN];                    // 8 KB
    __shared__ float s_ml[2][kSplit][16][2];       // 1 KB: (m,l) per split/row
    __shared__ float s_acc[2][kSplit - 1][16][kF]; // 24 KB: partial outputs

    const int tid  = threadIdx.x;
    const int lane = tid & 31;
    const int wave = tid >> 5;
    const int rbL  = wave >> 2;               // local row-block: 0/1
    const int s    = wave & 3;                // j-split index

    const int rb = blockIdx.x * 2 + rbL;      // 16-row block id, 0..1023
    const int b  = rb >> 7;                   // 128 row-blocks per batch
    const int i0 = (rb & 127) << 4;
    const int M  = lane & 15;
    const int hi = lane >> 4;

    for (int t = tid; t < kN / 4; t += 256)
        ((float4*)s_dst)[t] = ((const float4*)(dstv + (long)b * kN))[t];
    __syncthreads();

    const int   i      = i0 + M;
    const float sv     = srcv[(long)b * kN + i];
    const int*  adjRow = adj + ((long)b * kN + i) * kN;

    const int k0 = hi ? 16 : 0;   // first K-octet this lane owns (A layout)
    const int k1 = hi ? 24 : 8;   // second K-octet
    const int kb = hi ? 16 : 0;   // B-matrix K offset for this lane

    float m = -INFINITY;
    float l = 0.f;
    v8f acc0 = {}, acc1 = {}, acc2 = {}, acc3 = {};

    const int jbase = s * kJChunk;
    for (int j0 = jbase; j0 < jbase + kJChunk; j0 += 32) {
        // --- adjacency (non-temporal: stream-once, keep hT in L2) ----------
        alignas(16) int   av[16];
        alignas(16) float dv[16];
        *(v4i*)(av + 0)  = __builtin_nontemporal_load((const v4i*)(adjRow + j0 + k0));
        *(v4i*)(av + 4)  = __builtin_nontemporal_load((const v4i*)(adjRow + j0 + k0 + 4));
        *(v4i*)(av + 8)  = __builtin_nontemporal_load((const v4i*)(adjRow + j0 + k1));
        *(v4i*)(av + 12) = __builtin_nontemporal_load((const v4i*)(adjRow + j0 + k1 + 4));
        ((float4*)dv)[0] = *(const float4*)(s_dst + j0 + k0);
        ((float4*)dv)[1] = *(const float4*)(s_dst + j0 + k0 + 4);
        ((float4*)dv)[2] = *(const float4*)(s_dst + j0 + k1);
        ((float4*)dv)[3] = *(const float4*)(s_dst + j0 + k1 + 4);

        // --- masked leaky-relu scores (log2 domain) ------------------------
        float e[16];
#pragma unroll
        for (int k = 0; k < 16; ++k) {
            float v = sv + dv[k];
            v = (v > 0.f) ? v : kAlpha * v;       // lrelu commutes with *log2e
            e[k] = (av[k] > 0) ? v : kNegFill2;
        }

        float tmax = e[0];
#pragma unroll
        for (int k = 1; k < 16; ++k) tmax = fmaxf(tmax, e[k]);
        tmax = fmaxf(tmax, __shfl_xor(tmax, 16, 32));

        // --- rescale only when some row's max actually moved ---------------
        if (__any(tmax > m)) {
            const float mnew  = fmaxf(m, tmax);
            const float alpha = __builtin_amdgcn_exp2f(m - mnew);
            m = mnew;
            l *= alpha;
#pragma unroll
            for (int r = 0; r < 8; ++r) {
                const float sc = __shfl(alpha, r + 8 * hi, 32);
                acc0[r] *= sc; acc1[r] *= sc; acc2[r] *= sc; acc3[r] *= sc;
            }
        }

        float ps = 0.f;
        v16h P;
#pragma unroll
        for (int k = 0; k < 16; ++k) {
            float p = __builtin_amdgcn_exp2f(e[k] - m);
            ps += p;
            P[k] = (_Float16)p;
        }
        ps += __shfl_xor(ps, 16, 32);
        l += ps;

        // --- P(16x32) x hT-tile(32x16) -> 4 WMMAs --------------------------
        const _Float16* hbase = hT + ((long)b * kF + M) * kN + j0 + kb;
        const long gstride = (long)16 * kN;

        v16h B0 = *(const v16h*)(hbase);
        v16h B1 = *(const v16h*)(hbase + gstride);
        v16h B2 = *(const v16h*)(hbase + 2 * gstride);
        v16h B3 = *(const v16h*)(hbase + 3 * gstride);

        acc0 = __builtin_amdgcn_wmma_f32_16x16x32_f16(false, P, false, B0,
                                                      (short)0, acc0, false, false);
        acc1 = __builtin_amdgcn_wmma_f32_16x16x32_f16(false, P, false, B1,
                                                      (short)0, acc1, false, false);
        acc2 = __builtin_amdgcn_wmma_f32_16x16x32_f16(false, P, false, B2,
                                                      (short)0, acc2, false, false);
        acc3 = __builtin_amdgcn_wmma_f32_16x16x32_f16(false, P, false, B3,
                                                      (short)0, acc3, false, false);
    }

    // --- cross-split combine: out = (sum_s 2^(m_s-gm) * acc_s) / gl --------
    if (lane < 16) {
        s_ml[rbL][s][lane][0] = m;
        s_ml[rbL][s][lane][1] = l;
    }
    __syncthreads();

    float gm = s_ml[rbL][0][M][0];
#pragma unroll
    for (int t = 1; t < kSplit; ++t) gm = fmaxf(gm, s_ml[rbL][t][M][0]);
    float gl = 0.f;
#pragma unroll
    for (int t = 0; t < kSplit; ++t)
        gl += s_ml[rbL][t][M][1] * __builtin_amdgcn_exp2f(s_ml[rbL][t][M][0] - gm);

    const float cfac = __builtin_amdgcn_exp2f(m - gm) * __builtin_amdgcn_rcpf(gl);
    float sc[8];
#pragma unroll
    for (int r = 0; r < 8; ++r)
        sc[r] = __shfl(cfac, r + 8 * hi, 32);   // per-C/D-row scale

    if (s != 0) {
#pragma unroll
        for (int r = 0; r < 8; ++r) {
            const int row = r + 8 * hi;
            float* p = &s_acc[rbL][s - 1][row][M];
            p[0]  = acc0[r] * sc[r];
            p[16] = acc1[r] * sc[r];
            p[32] = acc2[r] * sc[r];
            p[48] = acc3[r] * sc[r];
        }
    }
    __syncthreads();

    if (s == 0) {
#pragma unroll
        for (int r = 0; r < 8; ++r) {
            const int row = r + 8 * hi;
            const float* p0 = &s_acc[rbL][0][row][M];
            const float* p1 = &s_acc[rbL][1][row][M];
            const float* p2 = &s_acc[rbL][2][row][M];
            float* orow = out + ((long)b * kN + i0 + row) * kF + M;
            orow[0]  = acc0[r] * sc[r] + p0[0]  + p1[0]  + p2[0];
            orow[16] = acc1[r] * sc[r] + p0[16] + p1[16] + p2[16];
            orow[32] = acc2[r] * sc[r] + p0[32] + p1[32] + p2[32];
            orow[48] = acc3[r] * sc[r] + p0[48] + p1[48] + p2[48];
        }
    }
}

// ---------------------------------------------------------------------------
extern "C" void kernel_launch(void* const* d_in, const int* in_sizes, int n_in,
                              void* d_out, int out_size, void* d_ws, size_t ws_size,
                              hipStream_t stream) {
    (void)in_sizes; (void)n_in; (void)out_size; (void)ws_size;

    const float* x   = (const float*)d_in[0];   // [B,N,64]
    const int*   adj = (const int*)  d_in[1];   // [B,N,N]
    const float* W   = (const float*)d_in[2];   // [64,64]
    const float* a   = (const float*)d_in[3];   // [128]
    float*       out = (float*)d_out;           // [B,N,64]

    char* ws = (char*)d_ws;
    _Float16* hT   = (_Float16*)ws;                             // 2 MB
    float*    srcv = (float*)(ws + (size_t)kB * kF * kN * 2);   // 64 KB
    float*    dstv = srcv + (size_t)kB * kN;                    // 64 KB

    gat_proj_kernel<<<kB * kN / 256, 256, 0, stream>>>(x, W, a, hT, srcv, dstv);
    // 1024 row-blocks x 4 split-waves / 8 waves per block = 512 blocks
    gat_attn_kernel<<<kB * kN / 16 / 2, 256, 0, stream>>>(adj, hT, srcv, dstv, out);
}
